// SampleModel_41034117546384
// MI455X (gfx1250) — matrix operand order repaired and checked by reading, not compiled, not run on hardware
//
#include <hip/hip_runtime.h>
#include <math.h>

#define B_ROWS 4096
#define K_CENT 8192
#define D_DIM  256
#define INVT   (1.0f / 0.07f)
#define LDSP   264   // LDS pitch in bf16 elems: 264*2B=528B -> 4-bank rotation per row

#define KSPLIT 4     // GEMM1 split-K  (4 * 256 blocks  = 1024 waves)
#define MSPLIT 2     // GEMM2 split-M  (2 * 512 blocks  = 1024 waves)

typedef __bf16 v16bf __attribute__((ext_vector_type(16)));
typedef __bf16 v8bf  __attribute__((ext_vector_type(8)));
typedef float  v8f   __attribute__((ext_vector_type(8)));

static __device__ inline v16bf mk16(v8bf lo, v8bf hi) {
    v16bf r;
#pragma unroll
    for (int i = 0; i < 8; ++i) { r[i] = lo[i]; r[i + 8] = hi[i]; }
    return r;
}

static __device__ inline v8f wmma_bf16(v16bf a, v16bf b, v8f c) {
    // D = A(16x32 bf16) * B(32x16 bf16) + C(16x16 f32)
    return __builtin_amdgcn_wmma_f32_16x16x32_bf16(false, a, false, b, (short)0, c, false, false);
}

// ---------------------------------------------------------------------------
// Kernel 1: (optionally L2-normalize rows) and split f32 -> bf16 hi + bf16 lo
// One wave (32 lanes) per row of 256 floats; 8 waves per 256-thread block.
// ---------------------------------------------------------------------------
__global__ void k_split(const float* __restrict__ src,
                        __bf16* __restrict__ hi, __bf16* __restrict__ lo,
                        int rows, int normalize) {
    int lane = threadIdx.x & 31;
    int row  = blockIdx.x * 8 + (threadIdx.x >> 5);
    if (row >= rows) return;
    const float* p = src + (size_t)row * D_DIM;
    float x[8];
    float ss = 0.0f;
#pragma unroll
    for (int j = 0; j < 8; ++j) {
        x[j] = p[lane + j * 32];
        ss += x[j] * x[j];
    }
    float inv = 1.0f;
    if (normalize) {
#pragma unroll
        for (int m = 16; m >= 1; m >>= 1) ss += __shfl_xor(ss, m, 32);
        inv = 1.0f / fmaxf(sqrtf(ss), 1e-12f);
    }
#pragma unroll
    for (int j = 0; j < 8; ++j) {
        float y = x[j] * inv;
        __bf16 h = (__bf16)y;
        __bf16 l = (__bf16)(y - (float)h);
        hi[(size_t)row * D_DIM + lane + j * 32] = h;
        lo[(size_t)row * D_DIM + lane + j * 32] = l;
    }
}

// ---------------------------------------------------------------------------
// Kernel 2: prod = F . c^T / T  with fused per-row max + argmax, split-K.
// blockIdx.x = M tile (16 feature rows), blockIdx.y = K split (2048 centroids).
// One wave per block; A tile staged in LDS (compiler hoists it into VGPRs).
// Split-bf16 product hi*hi + hi*lo + lo*hi in THREE independent accumulators
// to break the WMMA->WMMA RAW chain on C.
// ---------------------------------------------------------------------------
__global__ __launch_bounds__(32)
void k_gemm1_argmax(const __bf16* __restrict__ fh, const __bf16* __restrict__ fl,
                    const __bf16* __restrict__ ch, const __bf16* __restrict__ cl,
                    float* __restrict__ rowmax_p, int* __restrict__ rowarg_p) {
    __shared__ __bf16 sAh[16 * LDSP];
    __shared__ __bf16 sAl[16 * LDSP];
    const int lane  = threadIdx.x;
    const int row0  = blockIdx.x * 16;
    const int split = blockIdx.y;
    const int kt0   = split * (K_CENT / 16 / KSPLIT);
    const int kt1   = kt0 + (K_CENT / 16 / KSPLIT);

    // Stage A tile (16 rows x 256) hi/lo into LDS, 128-bit chunks.
#pragma unroll
    for (int t = 0; t < 16; ++t) {
        int ci = t * 32 + lane;        // chunk of 8 bf16
        int r  = ci >> 5;
        int c  = (ci & 31) * 8;
        *(v8bf*)&sAh[r * LDSP + c] = *(const v8bf*)&fh[(size_t)(row0 + r) * D_DIM + c];
        *(v8bf*)&sAl[r * LDSP + c] = *(const v8bf*)&fl[(size_t)(row0 + r) * D_DIM + c];
    }
    __syncthreads();

    const int half = lane >> 4;   // 0: M/K-low half, 1: high half (M per ISA; K contiguous, this code's order)
    const int l16  = lane & 15;

    float best[8];
    int   bidx[8];
#pragma unroll
    for (int v = 0; v < 8; ++v) { best[v] = -3.4e38f; bidx[v] = 0x7fffffff; }

    for (int kt = kt0; kt < kt1; ++kt) {
        const int n = kt * 16 + l16;                            // centroid (column)
        const __bf16* bh = ch + (size_t)n * D_DIM + half * 16;  // B: lanes>=16 hold K=16..31
        const __bf16* bl = cl + (size_t)n * D_DIM + half * 16;
        v8f acc_hh = {0.f, 0.f, 0.f, 0.f, 0.f, 0.f, 0.f, 0.f};
        v8f acc_hl = {0.f, 0.f, 0.f, 0.f, 0.f, 0.f, 0.f, 0.f};
        v8f acc_lh = {0.f, 0.f, 0.f, 0.f, 0.f, 0.f, 0.f, 0.f};
#pragma unroll
        for (int d = 0; d < 8; ++d) {
            const int d0   = d * 32;
            const int aoff = l16 * LDSP + d0 + half * 16;  // A: lanes>=16 hold K=16..31
            v16bf Ah = mk16(*(v8bf*)&sAh[aoff], *(v8bf*)&sAh[aoff + 8]);
            v16bf Al = mk16(*(v8bf*)&sAl[aoff], *(v8bf*)&sAl[aoff + 8]);
            v16bf Bh = mk16(*(const v8bf*)&bh[d0], *(const v8bf*)&bh[d0 + 8]);
            v16bf Bl = mk16(*(const v8bf*)&bl[d0], *(const v8bf*)&bl[d0 + 8]);
            acc_hh = wmma_bf16(Ah, Bh, acc_hh);   // three independent chains
            acc_hl = wmma_bf16(Ah, Bl, acc_hl);
            acc_lh = wmma_bf16(Al, Bh, acc_lh);
        }
        // Row-wise max+argmax: for VGPR v, row = (half ? v+8 : v); the 16 N
        // values of that row live in this lane's 16-lane group.
#pragma unroll
        for (int v = 0; v < 8; ++v) {
            float s   = (acc_hh[v] + acc_hl[v] + acc_lh[v]) * INVT;
            int   idx = n;
#pragma unroll
            for (int m = 8; m >= 1; m >>= 1) {
                float ov = __shfl_xor(s, m, 32);
                int   oi = __shfl_xor(idx, m, 32);
                if (ov > s || (ov == s && oi < idx)) { s = ov; idx = oi; }
            }
            if (s > best[v] || (s == best[v] && idx < bidx[v])) { best[v] = s; bidx[v] = idx; }
        }
    }
    if (l16 == 0) {
#pragma unroll
        for (int v = 0; v < 8; ++v) {
            rowmax_p[split * B_ROWS + row0 + half * 8 + v] = best[v];
            rowarg_p[split * B_ROWS + row0 + half * 8 + v] = bidx[v];
        }
    }
}

// ---------------------------------------------------------------------------
// Kernel 3: S[k] = sum_j exp( (c_j . c_k) / T ), split-M.
// blockIdx.x = N tile (16 columns, staged in LDS), blockIdx.y = M split.
// ---------------------------------------------------------------------------
__global__ __launch_bounds__(32)
void k_gemm2_expsum(const __bf16* __restrict__ ch, const __bf16* __restrict__ cl,
                    float* __restrict__ S_p) {
    __shared__ __bf16 sBh[16 * LDSP];
    __shared__ __bf16 sBl[16 * LDSP];
    const int lane  = threadIdx.x;
    const int n0    = blockIdx.x * 16;
    const int split = blockIdx.y;
    const int mt0   = split * (K_CENT / 16 / MSPLIT);
    const int mt1   = mt0 + (K_CENT / 16 / MSPLIT);

#pragma unroll
    for (int t = 0; t < 16; ++t) {
        int ci = t * 32 + lane;
        int r  = ci >> 5;
        int c  = (ci & 31) * 8;
        *(v8bf*)&sBh[r * LDSP + c] = *(const v8bf*)&ch[(size_t)(n0 + r) * D_DIM + c];
        *(v8bf*)&sBl[r * LDSP + c] = *(const v8bf*)&cl[(size_t)(n0 + r) * D_DIM + c];
    }
    __syncthreads();

    const int half = lane >> 4;
    const int l16  = lane & 15;
    float colsum = 0.0f;

    for (int mt = mt0; mt < mt1; ++mt) {
        const int m = mt * 16 + l16;                      // source row j
        const __bf16* ah = ch + (size_t)m * D_DIM + half * 16;
        const __bf16* al = cl + (size_t)m * D_DIM + half * 16;
        v8f acc_hh = {0.f, 0.f, 0.f, 0.f, 0.f, 0.f, 0.f, 0.f};
        v8f acc_hl = {0.f, 0.f, 0.f, 0.f, 0.f, 0.f, 0.f, 0.f};
        v8f acc_lh = {0.f, 0.f, 0.f, 0.f, 0.f, 0.f, 0.f, 0.f};
#pragma unroll
        for (int d = 0; d < 8; ++d) {
            const int d0   = d * 32;
            const int boff = l16 * LDSP + d0 + half * 16;
            v16bf Ah = mk16(*(const v8bf*)&ah[d0], *(const v8bf*)&ah[d0 + 8]);
            v16bf Al = mk16(*(const v8bf*)&al[d0], *(const v8bf*)&al[d0 + 8]);
            v16bf Bh = mk16(*(v8bf*)&sBh[boff], *(v8bf*)&sBh[boff + 8]);
            v16bf Bl = mk16(*(v8bf*)&sBl[boff], *(v8bf*)&sBl[boff + 8]);
            acc_hh = wmma_bf16(Ah, Bh, acc_hh);
            acc_hl = wmma_bf16(Ah, Bl, acc_hl);
            acc_lh = wmma_bf16(Al, Bh, acc_lh);
        }
#pragma unroll
        for (int v = 0; v < 8; ++v)
            colsum += expf((acc_hh[v] + acc_hl[v] + acc_lh[v]) * INVT);
    }
    // lane l and lane l^16 hold the same column N = l&15 (different M halves)
    colsum += __shfl_xor(colsum, 16, 32);
    if (half == 0) S_p[split * K_CENT + n0 + l16] = colsum;
}

// ---------------------------------------------------------------------------
// Kernel 4: merge split-K argmax partials + split-M column-sum partials,
// then loss = -mean( log(pep) - log(pep + S[argmax]) ), pep = exp(max)*w.
// Argmax merge is exact: split index ranges are disjoint & ascending, and
// within/between splits ties resolve to the smallest index (jnp.argmax).
// ---------------------------------------------------------------------------
__global__ void k_loss(const float* __restrict__ rowmax_p, const int* __restrict__ rowarg_p,
                       const float* __restrict__ w, const float* __restrict__ S_p,
                       float* __restrict__ out) {
    __shared__ float red[256];
    int tid = threadIdx.x;
    float acc = 0.0f;
    for (int r = tid; r < B_ROWS; r += 256) {
        float best = -3.4e38f;
        int   bidx = 0x7fffffff;
#pragma unroll
        for (int s = 0; s < KSPLIT; ++s) {
            float v = rowmax_p[s * B_ROWS + r];
            int   i = rowarg_p[s * B_ROWS + r];
            if (v > best || (v == best && i < bidx)) { best = v; bidx = i; }
        }
        float ssum = 0.0f;
#pragma unroll
        for (int s = 0; s < MSPLIT; ++s) ssum += S_p[s * K_CENT + bidx];
        float pep = expf(best) * w[r];
        acc += logf(pep) - logf(pep + ssum);
    }
    red[tid] = acc;
    __syncthreads();
    for (int st = 128; st > 0; st >>= 1) {
        if (tid < st) red[tid] += red[tid + st];
        __syncthreads();
    }
    if (tid == 0) out[0] = -red[0] / (float)B_ROWS;
}

// ---------------------------------------------------------------------------
extern "C" void kernel_launch(void* const* d_in, const int* in_sizes, int n_in,
                              void* d_out, int out_size, void* d_ws, size_t ws_size,
                              hipStream_t stream) {
    const float* features  = (const float*)d_in[0];   // [4096, 256]
    const float* centroids = (const float*)d_in[1];   // [8192, 256]
    const float* weight    = (const float*)d_in[2];   // [4096]

    char* w = (char*)d_ws;
    __bf16* c_hi = (__bf16*)w; w += (size_t)K_CENT * D_DIM * 2;
    __bf16* c_lo = (__bf16*)w; w += (size_t)K_CENT * D_DIM * 2;
    __bf16* f_hi = (__bf16*)w; w += (size_t)B_ROWS * D_DIM * 2;
    __bf16* f_lo = (__bf16*)w; w += (size_t)B_ROWS * D_DIM * 2;
    float*  rowmax_p = (float*)w; w += (size_t)KSPLIT * B_ROWS * 4;
    int*    rowarg_p = (int*)w;   w += (size_t)KSPLIT * B_ROWS * 4;
    float*  S_p      = (float*)w; w += (size_t)MSPLIT * K_CENT * 4;

    k_split<<<K_CENT / 8, 256, 0, stream>>>(centroids, c_hi, c_lo, K_CENT, 1);
    k_split<<<B_ROWS / 8, 256, 0, stream>>>(features,  f_hi, f_lo, B_ROWS, 0);

    dim3 g1(B_ROWS / 16, KSPLIT);
    k_gemm1_argmax<<<g1, 32, 0, stream>>>(f_hi, f_lo, c_hi, c_lo, rowmax_p, rowarg_p);

    dim3 g2(K_CENT / 16, MSPLIT);
    k_gemm2_expsum<<<g2, 32, 0, stream>>>(c_hi, c_lo, S_p);

    k_loss<<<1, 256, 0, stream>>>(rowmax_p, rowarg_p, weight, S_p, (float*)d_out);
}
